// GPT2SelfAttention_64209761076064
// MI455X (gfx1250) — matrix operand order repaired and checked by reading, not compiled, not run on hardware
//
#include <hip/hip_runtime.h>
#include <stdint.h>

#define BB 2
#define LL 2048
#define EE 1024
#define HH 16
#define HD 64

typedef __attribute__((ext_vector_type(2)))  float  v2f;
typedef __attribute__((ext_vector_type(8)))  float  v8f;
typedef __attribute__((ext_vector_type(16))) __bf16 v16bf;
typedef __attribute__((ext_vector_type(8)))  __bf16 v8bf;
typedef __attribute__((ext_vector_type(4)))  int    v4i;

union BFrag { v16bf v; v8bf h[2]; };

#if defined(__has_builtin)
#  if __has_builtin(__builtin_amdgcn_global_load_async_to_lds_b128) && __has_builtin(__builtin_amdgcn_s_wait_asynccnt)
#    define HAVE_ASYNC_LDS 1
#  endif
#endif

typedef __attribute__((address_space(1))) v4i g_v4i;   // global int4
typedef __attribute__((address_space(3))) v4i l_v4i;   // LDS int4

// Copy a contiguous block global -> LDS (bytes multiple of 16*nthreads; here 64KB/128thr).
__device__ __forceinline__ void copy_tile_to_lds(const float* gsrc, float* lds,
                                                 int bytes, int tid, int nthreads) {
#if defined(HAVE_ASYNC_LDS)
  const char* gb = (const char*)gsrc;
  char*       lb = (char*)lds;
  for (int off = tid * 16; off < bytes; off += nthreads * 16) {
    g_v4i* g = (g_v4i*)(uintptr_t)(gb + off);
    l_v4i* l = (l_v4i*)(uint32_t)(uintptr_t)(lb + off);
    __builtin_amdgcn_global_load_async_to_lds_b128(g, l, 0, 0);
  }
  __builtin_amdgcn_s_wait_asynccnt(0);
#else
  const float4* s = (const float4*)gsrc;
  float4*       d = (float4*)lds;
  for (int i = tid; i < bytes / 16; i += nthreads) d[i] = s[i];
#endif
}

// One 16x16 fp32 WMMA output tile: C = Alds[16xK] * W[KxN] (K = EE = 1024).
// A frag (16x4 f32): lane m (=lane&15) holds K = koff, koff+1 (koff = 0 for lanes<16, 2 else).
// B frag (4x16 f32): lane holds col n0; VGPR0/1 = rows k+koff, k+koff+1.
template <int N>
__device__ __forceinline__ v8f gemm_tile(const float* lds, const float* __restrict__ W,
                                         int n0, int koff, int row) {
  v8f c = {};
#pragma unroll 4
  for (int k = 0; k < EE; k += 4) {
    v2f a, b;
    a.x = lds[row * EE + k + koff];
    a.y = lds[row * EE + k + koff + 1];
    b.x = W[(size_t)(k + koff) * N + n0];
    b.y = W[(size_t)(k + koff + 1) * N + n0];
    c = __builtin_amdgcn_wmma_f32_16x16x4_f32(false, a, false, b, (short)0, c, false, false);
  }
  return c;
}

// Kernel 1: qkv = x @ W_in + b_in ; split heads; Q,K -> bf16, V -> f32.
__global__ __launch_bounds__(128) void k_qkv(const float* __restrict__ x,
                                             const float* __restrict__ Win,
                                             const float* __restrict__ bin,
                                             __bf16* __restrict__ Qb,
                                             __bf16* __restrict__ Kb,
                                             float* __restrict__ Vf) {
  extern __shared__ float lds[];  // 16 x 1024 f32 = 64KB
  const int tid = threadIdx.x, lane = tid & 31, wave = tid >> 5;
  const int m0 = blockIdx.x * 16;
  copy_tile_to_lds(x + (size_t)m0 * EE, lds, 16 * EE * 4, tid, 128);
  __syncthreads();

  const int ln = lane & 15;
  const int koff = (lane < 16) ? 0 : 2;
  const int n0 = blockIdx.y * 64 + wave * 16 + ln;
  v8f c = gemm_tile<3 * EE>(lds, Win, n0, koff, ln);

  const float bias = bin[n0];
  const int rbase = (lane < 16) ? 0 : 8;
#pragma unroll
  for (int r = 0; r < 8; ++r) {
    const int m = m0 + rbase + r;        // global row = b*L + l
    const int bI = m >> 11, l = m & (LL - 1);
    const float val = c[r] + bias;
    if (n0 < EE) {
      const int h = n0 >> 6, d = n0 & 63;
      Qb[(((size_t)bI * HH + h) * LL + l) * HD + d] = (__bf16)val;
    } else if (n0 < 2 * EE) {
      const int nn = n0 - EE, h = nn >> 6, d = nn & 63;
      Kb[(((size_t)bI * HH + h) * LL + l) * HD + d] = (__bf16)val;
    } else {
      const int nn = n0 - 2 * EE, h = nn >> 6, d = nn & 63;
      Vf[(((size_t)bI * HH + h) * LL + l) * HD + d] = val;
    }
  }
}

// Kernel 2: fused scores (bf16 WMMA) + scale + causal + mask + softmax + P@V (f32 WMMA).
// One WG per (b,h, 16-query-row tile). LDS holds the 16x2048 fp32 probability rows.
__global__ __launch_bounds__(128) void k_attn(const __bf16* __restrict__ Qb,
                                              const __bf16* __restrict__ Kb,
                                              const float* __restrict__ Vf,
                                              const float* __restrict__ mask,
                                              float* __restrict__ attn,
                                              float* __restrict__ ctx) {
  extern __shared__ float S[];        // 16*2048 + 192 floats
  float* red = S + 16 * LL;
  const int tid = threadIdx.x, lane = tid & 31, wave = tid >> 5;
  const int ln = lane & 15;
  const int bh = blockIdx.x >> 7;     // b*16 + h
  const int qt = blockIdx.x & 127;
  const int b = bh >> 4, h = bh & 15;
  const int q0 = qt * 16;
  const __bf16* Qbase = Qb + (size_t)bh * LL * HD;
  const __bf16* Kbase = Kb + (size_t)bh * LL * HD;
  const float*  Vbase = Vf + (size_t)bh * LL * HD;

  // Q A-fragments (16-bit A 16x32 layout): lane m holds K dsel..dsel+7 and 16+dsel..23+dsel.
  const int dsel = (lane < 16) ? 0 : 8;
  const int qr = q0 + ln;
  BFrag a0, a1;
  a0.h[0] = *(const v8bf*)(Qbase + qr * HD + dsel);
  a0.h[1] = *(const v8bf*)(Qbase + qr * HD + 16 + dsel);
  a1.h[0] = *(const v8bf*)(Qbase + qr * HD + 32 + dsel);
  a1.h[1] = *(const v8bf*)(Qbase + qr * HD + 48 + dsel);

  const int rbase = (lane < 16) ? 0 : 8;

  // Phase 1: S[16][2048] score tiles, 2 bf16 WMMAs (HD=64 = 2*K32) per 16x16 tile.
  for (int kt = wave; kt < LL / 16; kt += 4) {
    const int kcol = kt * 16 + ln;                    // key row / score column
    const int db = (lane < 16) ? 0 : 8;               // B 32x16 layout: lane holds K db..db+7 and 16+db..23+db
    const __bf16* krow = Kbase + (size_t)kcol * HD;
    BFrag b0, b1;
    b0.h[0] = *(const v8bf*)(krow + db);            // d 0..31 half
    b0.h[1] = *(const v8bf*)(krow + 16 + db);
    b1.h[0] = *(const v8bf*)(krow + 32 + db);       // d 32..63 half
    b1.h[1] = *(const v8bf*)(krow + 48 + db);
    v8f cc = {};
    cc = __builtin_amdgcn_wmma_f32_16x16x32_bf16(false, a0.v, false, b0.v, (short)0, cc, false, false);
    cc = __builtin_amdgcn_wmma_f32_16x16x32_bf16(false, a1.v, false, b1.v, (short)0, cc, false, false);
    const float madd = mask[b * LL + kcol];
#pragma unroll
    for (int r = 0; r < 8; ++r) {
      const int qg = q0 + rbase + r;
      float s = (float)(__bf16)cc[r];   // reference scores are bf16 before promotion
      s *= 0.125f;                      // 1/sqrt(64)
      if (kcol > qg) s = -10000.0f;     // causal
      S[(rbase + r) * LL + kcol] = s + madd;
    }
  }
  __syncthreads();

  // Phase 2: softmax over each of the 16 rows (8 threads x 256 cols per row).
  {
    const int r = tid >> 3, seg = tid & 7;
    const float* row = S + r * LL + seg * 256;
    float mx = -3.4028235e38f;
    for (int i = 0; i < 256; ++i) mx = fmaxf(mx, row[i]);
    red[tid] = mx;
  }
  __syncthreads();
  if (tid < 16) {
    float mx = red[tid * 8];
#pragma unroll
    for (int i = 1; i < 8; ++i) mx = fmaxf(mx, red[tid * 8 + i]);
    red[128 + tid] = mx;
  }
  __syncthreads();
  {
    const int r = tid >> 3, seg = tid & 7;
    const float mx = red[128 + r];
    float* row = S + r * LL + seg * 256;
    float sum = 0.f;
    for (int i = 0; i < 256; ++i) { float e = __expf(row[i] - mx); row[i] = e; sum += e; }
    red[tid] = sum;
  }
  __syncthreads();
  if (tid < 16) {
    float s = red[tid * 8];
#pragma unroll
    for (int i = 1; i < 8; ++i) s += red[tid * 8 + i];
    red[160 + tid] = 1.0f / s;
  }
  __syncthreads();
  // Normalize in LDS and stream attn to HBM once (coalesced).
  float* abase = attn + ((size_t)bh * LL + q0) * LL;
  for (int idx = tid; idx < 16 * LL; idx += 128) {
    const int r = idx >> 11;
    const float p = S[idx] * red[160 + r];
    S[idx] = p;
    abase[idx] = p;
  }
  __syncthreads();

  // Phase 3: ctx[16x64] = P[16x2048] @ V[2048x64], f32 WMMA, one 16-col tile per wave.
  const int koff = (lane < 16) ? 0 : 2;
  const int nc = wave * 16 + ln;  // 0..63
  v8f cc = {};
#pragma unroll 4
  for (int k = 0; k < LL; k += 4) {
    v2f a, bb;
    a.x = S[ln * LL + k + koff];
    a.y = S[ln * LL + k + koff + 1];
    bb.x = Vbase[(size_t)(k + koff) * HD + nc];
    bb.y = Vbase[(size_t)(k + koff + 1) * HD + nc];
    cc = __builtin_amdgcn_wmma_f32_16x16x4_f32(false, a, false, bb, (short)0, cc, false, false);
  }
#pragma unroll
  for (int r = 0; r < 8; ++r) {
    const int qg = q0 + rbase + r;
    ctx[((size_t)b * LL + qg) * EE + h * HD + nc] = cc[r];  // merged-head layout
  }
}

// Kernel 3: out = ctx @ W_out + b_out.
__global__ __launch_bounds__(128) void k_outproj(const float* __restrict__ ctx,
                                                 const float* __restrict__ Wout,
                                                 const float* __restrict__ bout,
                                                 float* __restrict__ out) {
  extern __shared__ float lds[];  // 16 x 1024 f32
  const int tid = threadIdx.x, lane = tid & 31, wave = tid >> 5;
  const int m0 = blockIdx.x * 16;
  copy_tile_to_lds(ctx + (size_t)m0 * EE, lds, 16 * EE * 4, tid, 128);
  __syncthreads();

  const int ln = lane & 15;
  const int koff = (lane < 16) ? 0 : 2;
  const int n0 = blockIdx.y * 64 + wave * 16 + ln;
  v8f c = gemm_tile<EE>(lds, Wout, n0, koff, ln);

  const float bias = bout[n0];
  const int rbase = (lane < 16) ? 0 : 8;
#pragma unroll
  for (int r = 0; r < 8; ++r) {
    const int m = m0 + rbase + r;
    out[(size_t)m * EE + n0] = c[r] + bias;
  }
}

extern "C" void kernel_launch(void* const* d_in, const int* in_sizes, int n_in,
                              void* d_out, int out_size, void* d_ws, size_t ws_size,
                              hipStream_t stream) {
  (void)in_sizes; (void)n_in; (void)out_size; (void)ws_size;
  const float* x    = (const float*)d_in[0];
  const float* mask = (const float*)d_in[1];
  const float* Win  = (const float*)d_in[2];
  const float* bin  = (const float*)d_in[3];
  const float* Wout = (const float*)d_in[4];
  const float* bout = (const float*)d_in[5];

  char* ws = (char*)d_ws;
  __bf16* Qb  = (__bf16*)(ws);                               //  8 MiB
  __bf16* Kb  = (__bf16*)(ws + (size_t)8  * 1024 * 1024);    //  8 MiB
  float*  Vf  = (float*) (ws + (size_t)16 * 1024 * 1024);    // 16 MiB
  float*  ctx = (float*) (ws + (size_t)32 * 1024 * 1024);    // 16 MiB

  float* out  = (float*)d_out;                               // [B,L,E]
  float* attn = (float*)d_out + (size_t)BB * LL * EE;        // [B,H,L,L]

  dim3 g1(256, 48);  // (M/16) x (3E/64)
  k_qkv<<<g1, 128, 16 * EE * sizeof(float), stream>>>(x, Win, bin, Qb, Kb, Vf);

  k_attn<<<BB * HH * (LL / 16), 128, (16 * LL + 192) * sizeof(float), stream>>>(
      Qb, Kb, Vf, mask, attn, ctx);

  dim3 g3(256, 16);  // (M/16) x (E/64)
  k_outproj<<<g3, 128, 16 * EE * sizeof(float), stream>>>(ctx, Wout, bout, out);
}
